// Model_4879082848288
// MI455X (gfx1250) — matrix operand order repaired and checked by reading, not compiled, not run on hardware
//
#include <hip/hip_runtime.h>
#include <hip/hip_bf16.h>

// ---------------------------------------------------------------------------
// CDNA5 / gfx1250 implementation.  Heavy GEMMs (projection + residual convs)
// use v_wmma_f32_16x16x32_f16 with f32 accumulation; activations NHWC f16.
// ---------------------------------------------------------------------------

typedef _Float16 half_t;
typedef __attribute__((ext_vector_type(16))) _Float16 v16h;
typedef __attribute__((ext_vector_type(8)))  _Float16 v8h;
typedef __attribute__((ext_vector_type(8)))  float    v8f;

union AFrag { v16h v; v8h h[2]; };

#define BN_INV_C 0.99999500003749981f  // 1/sqrt(1+1e-5)

__device__ __forceinline__ float selu_f(float x) {
  const float a = 1.6732632423543772f, s = 1.0507009873554805f;
  return x > 0.f ? s * x : s * a * (__expf(x) - 1.f);
}
__device__ __forceinline__ v8f zero_v8f() {
  v8f z;
#pragma unroll
  for (int i = 0; i < 8; ++i) z[i] = 0.f;
  return z;
}
__device__ __forceinline__ v8h zero_v8h() {
  v8h z;
#pragma unroll
  for (int i = 0; i < 8; ++i) z[i] = (_Float16)0.f;
  return z;
}
__device__ __forceinline__ v8f wmma16(v16h a, v16h b, v8f c) {
  return __builtin_amdgcn_wmma_f32_16x16x32_f16(false, a, false, b, (short)0, c,
                                                false, false);
}

// ---------------------------------------------------------------------------
// Projection GEMM: P[M,128] = X[M,768] @ W[768,128] + bias   (f32 in, f16 out)
// 256 threads = 8 waves; block tile 128(M) x 128(N).
// Double-buffered LDS; fragments preloaded so the 8 WMMAs issue back-to-back.
// ---------------------------------------------------------------------------
__global__ void __launch_bounds__(256) k_proj(const float* __restrict__ X,
                                              const float* __restrict__ W,
                                              const float* __restrict__ bias,
                                              half_t* __restrict__ P, int M) {
  __shared__ __align__(16) half_t lds_a[2][128 * 32];
  __shared__ __align__(16) half_t lds_b[2][128 * 32];  // transposed [n][k]
  const int tid = threadIdx.x;
  const int wave = tid >> 5, lane = tid & 31;
  const int hi = lane >> 4, lm = lane & 15;
  const int row0 = blockIdx.x * 128;
  v8f acc[8];
#pragma unroll
  for (int i = 0; i < 8; ++i) acc[i] = zero_v8f();

  // per-thread staging coordinates (constant across chunks)
  int ar[4], ak[4], bk[4], bn[4];
#pragma unroll
  for (int it = 0; it < 4; ++it) {
    int g = tid + it * 256;
    ar[it] = g >> 3;  ak[it] = (g & 7) << 2;   // A: row, k-quad
    bk[it] = g >> 5;  bn[it] = (g & 31) << 2;  // B: k, n-quad
  }
  float4 ra[4], rb[4];

  // prologue: stage chunk 0 into buffer 0
#pragma unroll
  for (int it = 0; it < 4; ++it) {
    float4 v = {0.f, 0.f, 0.f, 0.f};
    if (row0 + ar[it] < M)
      v = *(const float4*)(X + (size_t)(row0 + ar[it]) * 768 + ak[it]);
    ra[it] = v;
    rb[it] = *(const float4*)(W + (size_t)bk[it] * 128 + bn[it]);
  }
#pragma unroll
  for (int it = 0; it < 4; ++it) {
    half_t* pa = &lds_a[0][ar[it] * 32 + ak[it]];
    pa[0] = (half_t)ra[it].x; pa[1] = (half_t)ra[it].y;
    pa[2] = (half_t)ra[it].z; pa[3] = (half_t)ra[it].w;
    lds_b[0][(bn[it] + 0) * 32 + bk[it]] = (half_t)rb[it].x;
    lds_b[0][(bn[it] + 1) * 32 + bk[it]] = (half_t)rb[it].y;
    lds_b[0][(bn[it] + 2) * 32 + bk[it]] = (half_t)rb[it].z;
    lds_b[0][(bn[it] + 3) * 32 + bk[it]] = (half_t)rb[it].w;
  }
  __syncthreads();

  for (int ch = 0; ch < 24; ++ch) {
    const int cb = ch & 1, nb = cb ^ 1;
    if (ch + 1 < 24) {  // issue global loads for next chunk under the WMMAs
      const int k0n = (ch + 1) * 32;
#pragma unroll
      for (int it = 0; it < 4; ++it) {
        float4 v = {0.f, 0.f, 0.f, 0.f};
        if (row0 + ar[it] < M)
          v = *(const float4*)(X + (size_t)(row0 + ar[it]) * 768 + k0n + ak[it]);
        ra[it] = v;
        rb[it] = *(const float4*)(W + (size_t)(k0n + bk[it]) * 128 + bn[it]);
      }
      if (ch + 2 < 24 && row0 + (tid & 127) < M)
        __builtin_prefetch(X + (size_t)(row0 + (tid & 127)) * 768 + (ch + 2) * 32,
                           0, 0);
    }
    // fragment loads (all DS loads issued together), then 8 WMMAs back-to-back
    AFrag af;
    const int mr = wave * 16 + lm;
    af.h[0] = *(const v8h*)&lds_a[cb][mr * 32 + hi * 8];
    af.h[1] = *(const v8h*)&lds_a[cb][mr * 32 + 16 + hi * 8];
    AFrag bf[8];
#pragma unroll
    for (int nt = 0; nt < 8; ++nt) {
      const half_t* bp = &lds_b[cb][(nt * 16 + lm) * 32 + hi * 8];
      bf[nt].h[0] = *(const v8h*)bp;
      bf[nt].h[1] = *(const v8h*)(bp + 16);
    }
#pragma unroll
    for (int nt = 0; nt < 8; ++nt) acc[nt] = wmma16(af.v, bf[nt].v, acc[nt]);

    if (ch + 1 < 24) {  // store staged regs into the other buffer
#pragma unroll
      for (int it = 0; it < 4; ++it) {
        half_t* pa = &lds_a[nb][ar[it] * 32 + ak[it]];
        pa[0] = (half_t)ra[it].x; pa[1] = (half_t)ra[it].y;
        pa[2] = (half_t)ra[it].z; pa[3] = (half_t)ra[it].w;
        lds_b[nb][(bn[it] + 0) * 32 + bk[it]] = (half_t)rb[it].x;
        lds_b[nb][(bn[it] + 1) * 32 + bk[it]] = (half_t)rb[it].y;
        lds_b[nb][(bn[it] + 2) * 32 + bk[it]] = (half_t)rb[it].z;
        lds_b[nb][(bn[it] + 3) * 32 + bk[it]] = (half_t)rb[it].w;
      }
    }
    __syncthreads();
  }
#pragma unroll
  for (int nt = 0; nt < 8; ++nt) {
    int col = nt * 16 + lm;
    float bs = bias[col];
#pragma unroll
    for (int r = 0; r < 8; ++r) {
      int row = row0 + wave * 16 + hi * 8 + r;
      if (row < M) P[(size_t)row * 128 + col] = (half_t)(acc[nt][r] + bs);
    }
  }
}

// ---------------------------------------------------------------------------
// Implicit-GEMM conv (NHWC f16, weights f16 [tap][co][ci]); Cin,Cout % 32/16.
// One wave computes 16 consecutive output w-positions x Cout channels.
// ---------------------------------------------------------------------------
__global__ void __launch_bounds__(256) k_conv_wmma(
    const half_t* __restrict__ in, half_t* __restrict__ out,
    const half_t* __restrict__ wgt, const float* __restrict__ bias, int Bn,
    int Hin, int Win, int Cin, int Hout, int Wout, int Cout, int KH, int KW,
    int PH, int PW) {
  const int b = blockIdx.z;
  const int h = blockIdx.y;
  const int wave = threadIdx.x >> 5;
  const int lane = threadIdx.x & 31;
  const int w0 = blockIdx.x * 128 + wave * 16;
  if (w0 >= Wout) return;  // whole-wave uniform exit
  const int lm = lane & 15, hi = lane >> 4;
  const int nK = Cin >> 5;
  const int nN = Cout >> 4;  // <= 4
  v8f acc[4];
#pragma unroll
  for (int t = 0; t < 4; ++t) acc[t] = zero_v8f();

  const int wv_ = w0 + lm;  // this lane's A-row w position
  for (int dh = 0; dh < KH; ++dh) {
    int hin = h + dh - PH;
    if (hin < 0 || hin >= Hin) continue;
    for (int dw = 0; dw < KW; ++dw) {
      const int tap = dh * KW + dw;
      const int win = wv_ + dw - PW;
      const bool va = (win >= 0) && (win < Win);
      const half_t* arow = nullptr;
      if (va) arow = in + (((size_t)b * Hin + hin) * Win + win) * (size_t)Cin;
      for (int kc = 0; kc < nK; ++kc) {
        AFrag af;
        if (va) {
          af.h[0] = *(const v8h*)(arow + kc * 32 + hi * 8);
          af.h[1] = *(const v8h*)(arow + kc * 32 + 16 + hi * 8);
        } else {
          af.h[0] = zero_v8h();
          af.h[1] = zero_v8h();
        }
        AFrag bf[4];
#pragma unroll
        for (int nt = 0; nt < 4; ++nt) {
          if (nt < nN) {
            const half_t* bp =
                wgt + ((size_t)(tap * Cout + nt * 16 + lm)) * Cin + kc * 32 +
                hi * 8;
            bf[nt].h[0] = *(const v8h*)bp;
            bf[nt].h[1] = *(const v8h*)(bp + 16);
          }
        }
#pragma unroll
        for (int nt = 0; nt < 4; ++nt)
          if (nt < nN) acc[nt] = wmma16(af.v, bf[nt].v, acc[nt]);
      }
    }
  }
#pragma unroll
  for (int nt = 0; nt < 4; ++nt) {
    if (nt < nN) {
      int n = lm + nt * 16;
      float bs = bias[n];
#pragma unroll
      for (int r = 0; r < 8; ++r) {
        int wo = w0 + hi * 8 + r;
        if (wo < Wout)
          out[(((size_t)b * Hout + h) * Wout + wo) * Cout + n] =
              (half_t)(acc[nt][r] + bs);
      }
    }
  }
}

// Direct conv for Cin==1 (block 0 conv1 / down): tiny, VALU.
__global__ void k_conv_cin1(const half_t* __restrict__ in,
                            half_t* __restrict__ out,
                            const float* __restrict__ w,
                            const float* __restrict__ bias, int Bn, int Hin,
                            int Win, int Hout, int Wout, int Cout, int KH,
                            int KW, int PH, int PW) {
  long idx = (long)blockIdx.x * blockDim.x + threadIdx.x;
  long tot = (long)Bn * Hout * Wout * Cout;
  if (idx >= tot) return;
  int co = (int)(idx % Cout);
  long t = idx / Cout;
  int ww = (int)(t % Wout); t /= Wout;
  int h = (int)(t % Hout);
  int b = (int)(t / Hout);
  float acc = bias[co];
  for (int dh = 0; dh < KH; ++dh) {
    int hin = h + dh - PH;
    if (hin < 0 || hin >= Hin) continue;
    for (int dw = 0; dw < KW; ++dw) {
      int win = ww + dw - PW;
      if (win < 0 || win >= Win) continue;
      acc += (float)in[((long)b * Hin + hin) * Win + win] *
             w[(co * KH + dh) * KW + dw];
    }
  }
  out[idx] = (half_t)acc;
}

// OIHW f32 -> [tap][co][ci] f16
__global__ void k_convert_w(const float* __restrict__ w,
                            half_t* __restrict__ out, int Cout, int Cin,
                            int KH, int KW) {
  int idx = blockIdx.x * 256 + threadIdx.x;
  int tot = Cout * Cin * KH * KW;
  if (idx >= tot) return;
  int kw_ = idx % KW; int t = idx / KW;
  int kh_ = t % KH; t /= KH;
  int ci = t % Cin; int co = t / Cin;
  out[((size_t)((kh_ * KW + kw_) * Cout + co)) * Cin + ci] = (half_t)w[idx];
}

// ---------------------------------------------------------------------------
// Elementwise / pooling kernels
// ---------------------------------------------------------------------------
__global__ void k_pool33(const half_t* __restrict__ P, half_t* __restrict__ out,
                         const float* __restrict__ g,
                         const float* __restrict__ bb) {
  long idx = (long)blockIdx.x * blockDim.x + threadIdx.x;
  const long tot = 2L * 42 * 9477;
  if (idx >= tot) return;
  int ww = (int)(idx % 9477);
  long t = idx / 9477;
  int hh = (int)(t % 42);
  int b = (int)(t / 42);
  float mx = -1e30f;
  for (int j = 0; j < 3; ++j)
    for (int i = 0; i < 3; ++i) {
      float v = (float)P[((size_t)b * 28431 + (3 * ww + i)) * 128 + (3 * hh + j)];
      mx = fmaxf(mx, v);
    }
  out[idx] = (half_t)selu_f(mx * (g[0] * BN_INV_C) + bb[0]);
}

__global__ void k_bn_selu(const half_t* __restrict__ in,
                          half_t* __restrict__ out, long n, int C,
                          const float* __restrict__ g,
                          const float* __restrict__ bb) {
  long idx = (long)blockIdx.x * blockDim.x + threadIdx.x;
  if (idx >= n) return;
  int c = (int)(idx % C);
  out[idx] = (half_t)selu_f((float)in[idx] * (g[c] * BN_INV_C) + bb[c]);
}

__global__ void k_bn_selu_f32(const half_t* __restrict__ in,
                              float* __restrict__ out, long n, int C,
                              const float* __restrict__ g,
                              const float* __restrict__ bb) {
  long idx = (long)blockIdx.x * blockDim.x + threadIdx.x;
  if (idx >= n) return;
  int c = (int)(idx % C);
  out[idx] = selu_f((float)in[idx] * (g[c] * BN_INV_C) + bb[c]);
}

__global__ void k_add_pool3(const half_t* __restrict__ a,
                            const half_t* __restrict__ bsrc,
                            half_t* __restrict__ out, int Bn, int H, int Win,
                            int Wout, int C) {
  long idx = (long)blockIdx.x * blockDim.x + threadIdx.x;
  long tot = (long)Bn * H * Wout * C;
  if (idx >= tot) return;
  int c = (int)(idx % C);
  long t = idx / C;
  int w = (int)(t % Wout); t /= Wout;
  int h = (int)(t % H);
  int b = (int)(t / H);
  float mx = -1e30f;
  for (int i = 0; i < 3; ++i) {
    long src = (((long)b * H + h) * Win + (3 * w + i)) * C + c;
    mx = fmaxf(mx, (float)a[src] + (float)bsrc[src]);
  }
  out[idx] = (half_t)mx;
}

__global__ void k_add(float* __restrict__ y, const float* __restrict__ x,
                      int n) {
  int i = blockIdx.x * blockDim.x + threadIdx.x;
  if (i < n) y[i] += x[i];
}

// ---------------------------------------------------------------------------
// Attention pooling head (tiny tensors (B,42,13,64))
// ---------------------------------------------------------------------------
__global__ void k_att1(const float* __restrict__ hb, float* __restrict__ ab,
                       const float* __restrict__ w1,
                       const float* __restrict__ b1,
                       const float* __restrict__ g,
                       const float* __restrict__ bb) {
  long idx = (long)blockIdx.x * blockDim.x + threadIdx.x;
  long tot = 2L * 42 * 13 * 128;
  if (idx >= tot) return;
  int co = (int)(idx % 128);
  long pix = idx / 128;
  const float* src = hb + pix * 64;
  float acc = b1[co];
  for (int ci = 0; ci < 64; ++ci) acc += src[ci] * w1[co * 64 + ci];
  ab[idx] = selu_f(acc) * (g[co] * BN_INV_C) + bb[co];
}

__global__ void k_att2(const float* __restrict__ ab, float* __restrict__ wv,
                       const float* __restrict__ w2,
                       const float* __restrict__ b2) {
  long idx = (long)blockIdx.x * blockDim.x + threadIdx.x;
  long tot = 2L * 42 * 13 * 64;
  if (idx >= tot) return;
  int co = (int)(idx % 64);
  long pix = idx / 64;
  float acc = b2[co];
  for (int ci = 0; ci < 128; ++ci) acc += ab[pix * 128 + ci] * w2[co * 128 + ci];
  wv[idx] = acc;
}

__global__ void k_eS(const float* __restrict__ hb, const float* __restrict__ wv,
                     const float* __restrict__ pos, float* __restrict__ eS) {
  long idx = (long)blockIdx.x * blockDim.x + threadIdx.x;
  long tot = 2L * 42 * 64;
  if (idx >= tot) return;
  int c = (int)(idx % 64);
  long t = idx / 64;
  int h = (int)(t % 42);
  int b = (int)(t / 42);
  const float* wr = wv + (((long)b * 42 + h) * 13) * 64 + c;
  const float* hr = hb + (((long)b * 42 + h) * 13) * 64 + c;
  float mx = -1e30f;
  for (int w = 0; w < 13; ++w) mx = fmaxf(mx, wr[w * 64]);
  float sm = 0.f, acc = 0.f;
  for (int w = 0; w < 13; ++w) {
    float e = __expf(wr[w * 64] - mx);
    sm += e;
    acc += e * hr[w * 64];
  }
  acc /= sm;
  float coord = (h + 0.5f) * (64.f / 42.f) - 0.5f;
  coord = fminf(fmaxf(coord, 0.f), 63.f);
  int lo = (int)floorf(coord);
  int hi2 = lo + 1; if (hi2 > 63) hi2 = 63;
  float fw = coord - (float)lo;
  float pv = pos[lo * 64 + c] * (1.f - fw) + pos[hi2 * 64 + c] * fw;
  eS[idx] = acc + pv;
}

__global__ void k_eT(const float* __restrict__ hb, const float* __restrict__ wv,
                     float* __restrict__ eT) {
  long idx = (long)blockIdx.x * blockDim.x + threadIdx.x;
  long tot = 2L * 13 * 64;
  if (idx >= tot) return;
  int c = (int)(idx % 64);
  long t = idx / 64;
  int w = (int)(t % 13);
  int b = (int)(t / 13);
  const float* wr = wv + ((long)b * 42 * 13 + w) * 64 + c;
  const float* hr = hb + ((long)b * 42 * 13 + w) * 64 + c;
  const int str = 13 * 64;
  float mx = -1e30f;
  for (int h = 0; h < 42; ++h) mx = fmaxf(mx, wr[h * str]);
  float sm = 0.f, acc = 0.f;
  for (int h = 0; h < 42; ++h) {
    float e = __expf(wr[h * str] - mx);
    sm += e;
    acc += e * hr[h * str];
  }
  eT[idx] = acc / sm;
}

// ---------------------------------------------------------------------------
// GAT (D=64), one block per batch element
// ---------------------------------------------------------------------------
__global__ void __launch_bounds__(256) k_gat(
    const float* __restrict__ x, float* __restrict__ out, int N,
    const float* __restrict__ apw, const float* __restrict__ apb,
    const float* __restrict__ attw, const float* __restrict__ pwaw,
    const float* __restrict__ pwab, const float* __restrict__ pwow,
    const float* __restrict__ pwob, const float* __restrict__ bng,
    const float* __restrict__ bnb, float invtemp) {
  const int D = 64;
  __shared__ float xs[42 * 64];
  __shared__ float board[42 * 42];
  __shared__ float agg[42 * 64];
  int b = blockIdx.x, tid = threadIdx.x;
  for (int i = tid; i < N * D; i += 256) xs[i] = x[(size_t)b * N * D + i];
  __syncthreads();
  for (int p = tid; p < N * N; p += 256) {
    int i = p / N, j = p % N;
    float av[64];
#pragma unroll
    for (int o = 0; o < 64; ++o) av[o] = apb[o];
    for (int d = 0; d < 64; ++d) {
      float pd = xs[i * 64 + d] * xs[j * 64 + d];
#pragma unroll
      for (int o = 0; o < 64; ++o) av[o] += pd * apw[d * 64 + o];
    }
    float s = 0.f;
#pragma unroll
    for (int o = 0; o < 64; ++o) s += tanhf(av[o]) * attw[o];
    board[p] = s * invtemp;
  }
  __syncthreads();
  for (int i = tid; i < N; i += 256) {
    float mx = -1e30f;
    for (int j = 0; j < N; ++j) mx = fmaxf(mx, board[i * N + j]);
    float sm = 0.f;
    for (int j = 0; j < N; ++j) {
      float e = __expf(board[i * N + j] - mx);
      board[i * N + j] = e;
      sm += e;
    }
    float inv = 1.f / sm;
    for (int j = 0; j < N; ++j) board[i * N + j] *= inv;
  }
  __syncthreads();
  for (int p = tid; p < N * D; p += 256) {
    int i = p / D, d = p % D;
    float s = 0.f;
    for (int j = 0; j < N; ++j) s += board[i * N + j] * xs[j * D + d];
    agg[p] = s;
  }
  __syncthreads();
  for (int p = tid; p < N * D; p += 256) {
    int i = p / D, o = p % D;
    float hv = pwab[o] + pwob[o];
    for (int d = 0; d < D; ++d)
      hv += agg[i * D + d] * pwaw[d * D + o] + xs[i * D + d] * pwow[d * D + o];
    out[(size_t)b * N * D + p] =
        selu_f(hv * (bng[o] * BN_INV_C) + bnb[o]);
  }
}

// top-k pool (sorted desc, stable) ------------------------------------------
__global__ void k_pool_top(const float* __restrict__ x, float* __restrict__ out,
                           int N, int D, int keep, const float* __restrict__ pw,
                           const float* __restrict__ pb) {
  __shared__ float sc[64];
  int b = blockIdx.x, tid = threadIdx.x;
  for (int i = tid; i < N; i += blockDim.x) {
    float s = pb[0];
    for (int d = 0; d < D; ++d) s += x[((size_t)b * N + i) * D + d] * pw[d];
    sc[i] = 1.f / (1.f + __expf(-s));
  }
  __syncthreads();
  for (int i = tid; i < N; i += blockDim.x) {
    float si = sc[i];
    int rank = 0;
    for (int j = 0; j < N; ++j) {
      float sj = sc[j];
      if (sj > si || (sj == si && j < i)) ++rank;
    }
    if (rank < keep)
      for (int d = 0; d < D; ++d)
        out[((size_t)b * keep + rank) * D + d] = x[((size_t)b * N + i) * D + d] * si;
  }
}

// ---------------------------------------------------------------------------
// Heterogeneous-graph (HTRG) layer; one block per batch element. DO=32.
// ---------------------------------------------------------------------------
template <int DI>
__global__ void __launch_bounds__(256) k_htrg(
    const float* __restrict__ x1, int n1, const float* __restrict__ x2, int n2,
    const float* __restrict__ master, int mstride, const float* pt1w,
    const float* pt1b, const float* pt2w, const float* pt2b, const float* apw,
    const float* apb, const float* apMw, const float* apMb, const float* w11,
    const float* w22, const float* w12, const float* wM, const float* pwaw,
    const float* pwab, const float* pwow, const float* pwob,
    const float* pwaMw, const float* pwaMb, const float* pwoMw,
    const float* pwoMb, const float* bng, const float* bnb, float invtemp,
    float* __restrict__ outT, float* __restrict__ outS,
    float* __restrict__ outM) {
  const int DO = 32;
  __shared__ float xs[27 * 64];
  __shared__ float ms[64];
  __shared__ float board[27 * 27];
  __shared__ float am[27];
  __shared__ float agg[27 * 64];
  int b = blockIdx.x, tid = threadIdx.x;
  int N = n1 + n2;
  for (int p = tid; p < N * DI; p += 256) {
    int i = p / DI, d = p % DI;
    const float* xr; const float* wP; const float* bP;
    if (i < n1) { xr = x1 + ((size_t)b * n1 + i) * DI; wP = pt1w; bP = pt1b; }
    else        { xr = x2 + ((size_t)b * n2 + (i - n1)) * DI; wP = pt2w; bP = pt2b; }
    float s = bP[d];
    for (int k = 0; k < DI; ++k) s += xr[k] * wP[k * DI + d];
    xs[i * DI + d] = s;
  }
  if (tid < DI) ms[tid] = master[(size_t)b * mstride + tid];
  __syncthreads();
  for (int p = tid; p < N * N; p += 256) {
    int i = p / N, j = p % N;
    const float* wq = (i < n1) ? ((j < n1) ? w11 : w12)
                               : ((j < n1) ? w12 : w22);
    float av[DO];
#pragma unroll
    for (int o = 0; o < DO; ++o) av[o] = apb[o];
    for (int d = 0; d < DI; ++d) {
      float pd = xs[i * DI + d] * xs[j * DI + d];
#pragma unroll
      for (int o = 0; o < DO; ++o) av[o] += pd * apw[d * DO + o];
    }
    float s = 0.f;
#pragma unroll
    for (int o = 0; o < DO; ++o) s += tanhf(av[o]) * wq[o];
    board[p] = s * invtemp;
  }
  for (int j = tid; j < N; j += 256) {
    float av[DO];
#pragma unroll
    for (int o = 0; o < DO; ++o) av[o] = apMb[o];
    for (int d = 0; d < DI; ++d) {
      float pd = xs[j * DI + d] * ms[d];
#pragma unroll
      for (int o = 0; o < DO; ++o) av[o] += pd * apMw[d * DO + o];
    }
    float s = 0.f;
#pragma unroll
    for (int o = 0; o < DO; ++o) s += tanhf(av[o]) * wM[o];
    am[j] = s * invtemp;
  }
  __syncthreads();
  for (int i = tid; i < N; i += 256) {
    float mx = -1e30f;
    for (int j = 0; j < N; ++j) mx = fmaxf(mx, board[i * N + j]);
    float sm = 0.f;
    for (int j = 0; j < N; ++j) {
      float e = __expf(board[i * N + j] - mx);
      board[i * N + j] = e;
      sm += e;
    }
    float inv = 1.f / sm;
    for (int j = 0; j < N; ++j) board[i * N + j] *= inv;
  }
  if (tid == 0) {
    float mx = -1e30f;
    for (int j = 0; j < N; ++j) mx = fmaxf(mx, am[j]);
    float sm = 0.f;
    for (int j = 0; j < N; ++j) { float e = __expf(am[j] - mx); am[j] = e; sm += e; }
    float inv = 1.f / sm;
    for (int j = 0; j < N; ++j) am[j] *= inv;
  }
  __syncthreads();
  for (int p = tid; p < N * DI; p += 256) {
    int i = p / DI, d = p % DI;
    float s = 0.f;
    for (int j = 0; j < N; ++j) s += board[i * N + j] * xs[j * DI + d];
    agg[p] = s;
  }
  __syncthreads();
  for (int p = tid; p < N * DO; p += 256) {
    int i = p / DO, o = p % DO;
    float hv = pwab[o] + pwob[o];
    for (int d = 0; d < DI; ++d)
      hv += agg[i * DI + d] * pwaw[d * DO + o] + xs[i * DI + d] * pwow[d * DO + o];
    float y = selu_f(hv * (bng[o] * BN_INV_C) + bnb[o]);
    if (i < n1) outT[((size_t)b * n1 + i) * DO + o] = y;
    else        outS[((size_t)b * n2 + (i - n1)) * DO + o] = y;
  }
  __syncthreads();
  if (tid < DI) {
    float s = 0.f;
    for (int j = 0; j < N; ++j) s += am[j] * xs[j * DI + tid];
    agg[tid] = s;
  }
  __syncthreads();
  if (tid < DO) {
    float s = pwaMb[tid] + pwoMb[tid];
    for (int d = 0; d < DI; ++d)
      s += agg[d] * pwaMw[d * DO + tid] + ms[d] * pwoMw[d * DO + tid];
    outM[(size_t)b * DO + tid] = s;
  }
}

// final readout --------------------------------------------------------------
__global__ void k_final(const float* t1, const float* t2, int nT,
                        const float* s1, const float* s2, int nS,
                        const float* m1, const float* m2, const float* ow,
                        const float* ob, float* __restrict__ outp) {
  __shared__ float last[160];
  int b = blockIdx.x, tid = threadIdx.x;
  const int D = 32;
  if (tid < D) {
    int c = tid;
    float sum = 0.f, mab = -1e30f;
    for (int n_ = 0; n_ < nT; ++n_) {
      float v = fmaxf(t1[((size_t)b * nT + n_) * D + c],
                      t2[((size_t)b * nT + n_) * D + c]);
      mab = fmaxf(mab, fabsf(v));
      sum += v;
    }
    last[c] = mab;
    last[32 + c] = sum / (float)nT;
    sum = 0.f; mab = -1e30f;
    for (int n_ = 0; n_ < nS; ++n_) {
      float v = fmaxf(s1[((size_t)b * nS + n_) * D + c],
                      s2[((size_t)b * nS + n_) * D + c]);
      mab = fmaxf(mab, fabsf(v));
      sum += v;
    }
    last[64 + c] = mab;
    last[96 + c] = sum / (float)nS;
    last[128 + c] = fmaxf(m1[b * 32 + c], m2[b * 32 + c]);
  }
  __syncthreads();
  if (tid < 2) {
    float s = ob[tid];
    for (int i = 0; i < 160; ++i) s += last[i] * ow[i * 2 + tid];
    outp[b * 2 + tid] = s;
  }
}

// ---------------------------------------------------------------------------
// Host orchestration
// ---------------------------------------------------------------------------
static inline unsigned nbx(long long n) { return (unsigned)((n + 255) / 256); }

#define F(i) ((const float*)d_in[(i)])

extern "C" void kernel_launch(void* const* d_in, const int* in_sizes, int n_in,
                              void* d_out, int out_size, void* d_ws,
                              size_t ws_size, hipStream_t stream) {
  (void)in_sizes; (void)n_in; (void)out_size; (void)ws_size;
  const int Bn = 2, T = 28431;
  const int M = Bn * T;

  char* wsb = (char*)d_ws;
  size_t off = 0;
  auto alloc = [&](size_t bytes) -> char* {
    char* p = wsb + off;
    off = (off + bytes + 255) & ~(size_t)255;
    return p;
  };
  half_t* P    = (half_t*)alloc((size_t)M * 128 * 2);
  half_t* bufA = (half_t*)alloc(8491392ull * 2);
  half_t* bufB = (half_t*)alloc(8491392ull * 2);
  half_t* pre  = (half_t*)alloc(8491392ull * 2);
  half_t* t1   = (half_t*)alloc(26080704ull * 2);
  half_t* t2   = (half_t*)alloc(25474176ull * 2);
  half_t* idn  = (half_t*)alloc(25474176ull * 2);
  half_t* wcv  = (half_t*)alloc(262144ull * 2);
  float* hb   = (float*)alloc(69888ull * 4);
  float* abuf = (float*)alloc(139776ull * 4);
  float* wvb  = (float*)alloc(69888ull * 4);
  float* bES  = (float*)alloc(5376 * 4);
  float* bET  = (float*)alloc(1664 * 4);
  float* bGS  = (float*)alloc(5376 * 4);
  float* bGT  = (float*)alloc(1664 * 4);
  float* bOS  = (float*)alloc(2688 * 4);
  float* bOT  = (float*)alloc(768 * 4);
  float* hT1 = (float*)alloc(384 * 4);
  float* hS1 = (float*)alloc(1344 * 4);
  float* m1  = (float*)alloc(64 * 4);
  float* T1p = (float*)alloc(192 * 4);
  float* S1p = (float*)alloc(640 * 4);
  float* hT2 = (float*)alloc(384 * 4);
  float* hS2 = (float*)alloc(1344 * 4);
  float* m2  = (float*)alloc(64 * 4);
  float* T2p = (float*)alloc(192 * 4);
  float* S2p = (float*)alloc(640 * 4);
  float* tA = (float*)alloc(192 * 4);
  float* sA = (float*)alloc(640 * 4);
  float* mA = (float*)alloc(64 * 4);

  // ---- convert conv weights to f16 [tap][co][ci] ----
  struct CV { int wi, Cout, Cin, KH, KW; long offh; };
  const CV cvs[12] = {
      {11, 32, 32, 2, 3, 0},      {17, 32, 32, 2, 3, 6144},
      {21, 32, 32, 2, 3, 12288},  {25, 64, 32, 2, 3, 18432},
      {29, 64, 64, 2, 3, 30720},  {31, 64, 32, 1, 3, 55296},
      {35, 64, 64, 2, 3, 61440},  {39, 64, 64, 2, 3, 86016},
      {43, 64, 64, 2, 3, 110592}, {47, 64, 64, 2, 3, 135168},
      {51, 64, 64, 2, 3, 159744}, {55, 64, 64, 2, 3, 184320}};
  for (int i = 0; i < 12; ++i) {
    int tot = cvs[i].Cout * cvs[i].Cin * cvs[i].KH * cvs[i].KW;
    k_convert_w<<<dim3(nbx(tot)), dim3(256), 0, stream>>>(
        F(cvs[i].wi), wcv + cvs[i].offh, cvs[i].Cout, cvs[i].Cin, cvs[i].KH,
        cvs[i].KW);
  }

  // ---- projection GEMM + (3,3) maxpool + BN + SELU ----
  k_proj<<<dim3((M + 127) / 128), dim3(256), 0, stream>>>(
      (const float*)d_in[0], F(1), F(2), P, M);
  k_pool33<<<dim3(nbx((long)Bn * 42 * 9477)), dim3(256), 0, stream>>>(
      P, bufA, F(3), F(4));

  // ---- residual encoder blocks ----
  const int Wt[6] = {9477, 3159, 1053, 351, 117, 39};
  struct BlkI {
    int bn1g, bn1b, c1w, c1b, bn2g, bn2b, c2w, c2b, dw, db, Cin, Cout;
    long c1o, c2o, dno;
  };
  const BlkI blk[6] = {
      {-1, -1, 7, 8, 9, 10, 11, 12, 13, 14, 1, 32, -1, 0, -1},
      {15, 16, 17, 18, 19, 20, 21, 22, -1, -1, 32, 32, 6144, 12288, -1},
      {23, 24, 25, 26, 27, 28, 29, 30, 31, 32, 32, 64, 18432, 30720, 55296},
      {33, 34, 35, 36, 37, 38, 39, 40, -1, -1, 64, 64, 61440, 86016, -1},
      {41, 42, 43, 44, 45, 46, 47, 48, -1, -1, 64, 64, 110592, 135168, -1},
      {49, 50, 51, 52, 53, 54, 55, 56, -1, -1, 64, 64, 159744, 184320, -1}};

  half_t* cur = bufA;
  half_t* nxt = bufB;
  for (int i = 0; i < 6; ++i) {
    const BlkI& bk = blk[i];
    const int Win_ = Wt[i], Wout_ = Win_ / 3;
    const half_t* prep;
    if (i == 0) {
      prep = cur;
    } else {
      long nIn = (long)Bn * 42 * Win_ * bk.Cin;
      k_bn_selu<<<dim3(nbx(nIn)), dim3(256), 0, stream>>>(cur, pre, nIn, bk.Cin,
                                                          F(bk.bn1g), F(bk.bn1b));
      prep = pre;
    }
    // conv1: (B,42,Win,Cin) -> (B,43,Win,Cout), k(2,3) pad(1,1)
    if (bk.Cin == 1)
      k_conv_cin1<<<dim3(nbx((long)Bn * 43 * Win_ * bk.Cout)), dim3(256), 0,
                    stream>>>(prep, t1, F(bk.c1w), F(bk.c1b), Bn, 42, Win_, 43,
                              Win_, bk.Cout, 2, 3, 1, 1);
    else
      k_conv_wmma<<<dim3((Win_ + 127) / 128, 43, Bn), dim3(256), 0, stream>>>(
          prep, t1, wcv + bk.c1o, F(bk.c1b), Bn, 42, Win_, bk.Cin, 43, Win_,
          bk.Cout, 2, 3, 1, 1);
    long n1c = (long)Bn * 43 * Win_ * bk.Cout;
    k_bn_selu<<<dim3(nbx(n1c)), dim3(256), 0, stream>>>(t1, t1, n1c, bk.Cout,
                                                        F(bk.bn2g), F(bk.bn2b));
    // conv2: (B,43,Win,Cout) -> (B,42,Win,Cout), k(2,3) pad(0,1)
    k_conv_wmma<<<dim3((Win_ + 127) / 128, 42, Bn), dim3(256), 0, stream>>>(
        t1, t2, wcv + bk.c2o, F(bk.c2b), Bn, 43, Win_, bk.Cout, 42, Win_,
        bk.Cout, 2, 3, 0, 1);
    // identity / downsample path
    const half_t* idp;
    if (bk.dw >= 0) {
      if (bk.Cin == 1)
        k_conv_cin1<<<dim3(nbx((long)Bn * 42 * Win_ * bk.Cout)), dim3(256), 0,
                      stream>>>(cur, idn, F(bk.dw), F(bk.db), Bn, 42, Win_, 42,
                                Win_, bk.Cout, 1, 3, 0, 1);
      else
        k_conv_wmma<<<dim3((Win_ + 127) / 128, 42, Bn), dim3(256), 0, stream>>>(
            cur, idn, wcv + bk.dno, F(bk.db), Bn, 42, Win_, bk.Cin, 42, Win_,
            bk.Cout, 1, 3, 0, 1);
      idp = idn;
    } else {
      idp = cur;
    }
    k_add_pool3<<<dim3(nbx((long)Bn * 42 * Wout_ * bk.Cout)), dim3(256), 0,
                  stream>>>(t2, idp, nxt, Bn, 42, Win_, Wout_, bk.Cout);
    half_t* tmp = cur; cur = nxt; nxt = tmp;
  }

  // ---- attention pooling head ----
  long nHB = (long)Bn * 42 * 13 * 64;
  k_bn_selu_f32<<<dim3(nbx(nHB)), dim3(256), 0, stream>>>(cur, hb, nHB, 64,
                                                          F(5), F(6));
  k_att1<<<dim3(nbx((long)Bn * 42 * 13 * 128)), dim3(256), 0, stream>>>(
      hb, abuf, F(57), F(58), F(59), F(60));
  k_att2<<<dim3(nbx((long)Bn * 42 * 13 * 64)), dim3(256), 0, stream>>>(
      abuf, wvb, F(61), F(62));
  k_eS<<<dim3(nbx((long)Bn * 42 * 64)), dim3(256), 0, stream>>>(hb, wvb, F(63),
                                                                bES);
  k_eT<<<dim3(nbx((long)Bn * 13 * 64)), dim3(256), 0, stream>>>(hb, wvb, bET);

  // ---- GAT + pool ----
  k_gat<<<dim3(Bn), dim3(256), 0, stream>>>(bES, bGS, 42, F(66), F(67), F(68),
                                            F(69), F(70), F(71), F(72), F(73),
                                            F(74), 0.5f);
  k_gat<<<dim3(Bn), dim3(256), 0, stream>>>(bET, bGT, 13, F(75), F(76), F(77),
                                            F(78), F(79), F(80), F(81), F(82),
                                            F(83), 0.5f);
  k_pool_top<<<dim3(Bn), dim3(256), 0, stream>>>(bGS, bOS, 42, 64, 21, F(172),
                                                 F(173));
  k_pool_top<<<dim3(Bn), dim3(256), 0, stream>>>(bGT, bOT, 13, 64, 6, F(174),
                                                 F(175));

#define HTRG_ARGS(s)                                                          \
  F(s + 0), F(s + 1), F(s + 2), F(s + 3), F(s + 4), F(s + 5), F(s + 6),       \
      F(s + 7), F(s + 8), F(s + 9), F(s + 10), F(s + 11), F(s + 12),          \
      F(s + 13), F(s + 14), F(s + 15), F(s + 16), F(s + 17), F(s + 18),       \
      F(s + 19), F(s + 20), F(s + 21)

  // ---- branch 1 ----
  k_htrg<64><<<dim3(Bn), dim3(256), 0, stream>>>(bOT, 6, bOS, 21, F(64), 0,
                                                 HTRG_ARGS(84), 0.01f, hT1, hS1,
                                                 m1);
  k_pool_top<<<dim3(Bn), dim3(256), 0, stream>>>(hS1, S1p, 21, 32, 10, F(176),
                                                 F(177));
  k_pool_top<<<dim3(Bn), dim3(256), 0, stream>>>(hT1, T1p, 6, 32, 3, F(178),
                                                 F(179));
  k_htrg<32><<<dim3(Bn), dim3(256), 0, stream>>>(T1p, 3, S1p, 10, m1, 32,
                                                 HTRG_ARGS(106), 0.01f, tA, sA,
                                                 mA);
  k_add<<<dim3(1), dim3(256), 0, stream>>>(T1p, tA, Bn * 3 * 32);
  k_add<<<dim3(3), dim3(256), 0, stream>>>(S1p, sA, Bn * 10 * 32);
  k_add<<<dim3(1), dim3(64), 0, stream>>>(m1, mA, Bn * 32);

  // ---- branch 2 ----
  k_htrg<64><<<dim3(Bn), dim3(256), 0, stream>>>(bOT, 6, bOS, 21, F(65), 0,
                                                 HTRG_ARGS(128), 0.01f, hT2,
                                                 hS2, m2);
  k_pool_top<<<dim3(Bn), dim3(256), 0, stream>>>(hS2, S2p, 21, 32, 10, F(180),
                                                 F(181));
  k_pool_top<<<dim3(Bn), dim3(256), 0, stream>>>(hT2, T2p, 6, 32, 3, F(182),
                                                 F(183));
  k_htrg<32><<<dim3(Bn), dim3(256), 0, stream>>>(T2p, 3, S2p, 10, m2, 32,
                                                 HTRG_ARGS(150), 0.01f, tA, sA,
                                                 mA);
  k_add<<<dim3(1), dim3(256), 0, stream>>>(T2p, tA, Bn * 3 * 32);
  k_add<<<dim3(3), dim3(256), 0, stream>>>(S2p, sA, Bn * 10 * 32);
  k_add<<<dim3(1), dim3(64), 0, stream>>>(m2, mA, Bn * 32);

  // ---- final readout ----
  k_final<<<dim3(Bn), dim3(64), 0, stream>>>(T1p, T2p, 3, S1p, S2p, 10, m1, m2,
                                             F(184), F(185), (float*)d_out);
}
